// RegionProposalNetwork_35923106463955
// MI455X (gfx1250) — matrix operand order repaired and checked by reading, not compiled, not run on hardware
//
#include <hip/hip_runtime.h>
#include <hip/hip_bf16.h>
#include <math.h>

typedef __attribute__((ext_vector_type(16))) _Float16 v16h;
typedef __attribute__((ext_vector_type(8)))  float    v8f;
typedef unsigned long long u64;
typedef unsigned int u32;

#define FEAT_C   256
#define GRID_H   128
#define GRID_W   128
#define NPIX     (GRID_H * GRID_W)          // 16384
#define PADW     130                        // halo-padded width/height
#define NPPIX    (PADW * PADW)              // 16900
#define ANUM     9
#define N_ANCH   (NPIX * ANUM)              // 147456
#define SORT_N   262144                     // 2^18 >= N_ANCH
#define PRENMS   6000
#define SORT2_N  8192                       // 2^13 >= PRENMS
#define POSTK    300
#define IM_SZ    1024.0f
#define STRIDE_F 8.0f
#define NMS_THRV 0.7f
#define MIN_SZ   16.0f
#define DW_CLIPV 4.135166556742356f         // log(1000/16)

union AFrag { v16h v; _Float16 e[16]; };
union CFrag { v8f  v; float    e[8];  };

// Feature-detect gfx1250 async global->LDS copy builtins (device pass only).
#if defined(__gfx1250__) && defined(__has_builtin)
#if __has_builtin(__builtin_amdgcn_global_load_async_to_lds_b128)
#define USE_ASYNC_LDS 1
#endif
#endif

#ifdef USE_ASYNC_LDS
typedef int i4 __attribute__((vector_size(16)));
typedef __attribute__((address_space(1))) i4 gi4;   // global 16B chunk
typedef __attribute__((address_space(3))) i4 li4;   // LDS 16B chunk
#endif

__device__ __forceinline__ void copy16_g2l(const _Float16* src, _Float16* dst) {
#ifdef USE_ASYNC_LDS
    __builtin_amdgcn_global_load_async_to_lds_b128(
        (gi4*)(size_t)src, (li4*)(unsigned int)(size_t)dst, 0, 0);
#else
    *(uint4*)dst = *(const uint4*)src;
#endif
}

__device__ __forceinline__ void async_fence() {
#ifdef USE_ASYNC_LDS
#if __has_builtin(__builtin_amdgcn_s_wait_asynccnt)
    __builtin_amdgcn_s_wait_asynccnt(0);
#else
    asm volatile("s_wait_asynccnt 0x0" ::: "memory");
#endif
#endif
}

// ---------------------------------------------------------------------------
// Kernel 0: feat NCHW f32 -> halo-padded NHWC f16  pf[gy][gx][c], gy/gx in [0,130)
// ---------------------------------------------------------------------------
__global__ void feat_pad_kernel(const float* __restrict__ feat, _Float16* __restrict__ pf) {
    int idx = blockIdx.x * blockDim.x + threadIdx.x;
    if (idx >= NPPIX * FEAT_C) return;
    int c = idx & 255;
    int p = idx >> 8;
    int gx = p % PADW, gy = p / PADW;
    int x = gx - 1, y = gy - 1;
    float v = 0.f;
    if (x >= 0 && x < GRID_W && y >= 0 && y < GRID_H)
        v = feat[c * NPIX + y * GRID_W + x];
    pf[idx] = (_Float16)v;
}

// ---------------------------------------------------------------------------
// Kernel 1: repack conv weights OIHW fp32 -> Wt[tap][c_out][c_in] f16
// ---------------------------------------------------------------------------
__global__ void rpn_wrepack_kernel(const float* __restrict__ w, _Float16* __restrict__ Wt) {
    int l = blockIdx.x * blockDim.x + threadIdx.x;
    if (l >= 9 * 256 * 256) return;
    int ci  = l & 255;
    int co  = (l >> 8) & 255;
    int tap = l >> 16;                      // 0..8
    int kh = tap / 3, kw = tap % 3;
    Wt[l] = (_Float16)w[((co * 256 + ci) * 3 + kh) * 3 + kw];
}

// ---------------------------------------------------------------------------
// Kernel 2: 3x3 conv via WMMA f16. Block = 256 thr (8 waves) = 16-pixel x-strip
// x 256 output channels. Wave w owns channels [w*32, w*32+32): 2 accumulators.
// A (16x32, shared by all waves) and B (32x256) staged in LDS via async copies.
// ---------------------------------------------------------------------------
__global__ void rpn_conv_wmma_kernel(const _Float16* __restrict__ pf,
                                     const _Float16* __restrict__ Wt,
                                     const float* __restrict__ bias,
                                     _Float16* __restrict__ rf) {
    __shared__ _Float16 As[16 * 32];        // [m][k]  1 KB
    __shared__ _Float16 Bs[256 * 32];       // [n][k] 16 KB

    const int tid  = threadIdx.x;
    const int lane = tid & 31;
    const int wave = tid >> 5;
    const int half = lane >> 4;
    const int lm   = lane & 15;

    const int strip = blockIdx.x;           // 0..1023
    const int y     = strip >> 3;
    const int x0    = (strip & 7) << 4;
    const int n0    = wave * 32;

    CFrag acc0, acc1;
    acc0.v = (v8f){0.f, 0.f, 0.f, 0.f, 0.f, 0.f, 0.f, 0.f};
    acc1.v = acc0.v;

    for (int tap = 0; tap < 9; ++tap) {
        const int dy = tap / 3 - 1, dx = tap % 3 - 1;
        const int gy = y + dy + 1;          // padded coords, always in-bounds
        const int gxb = x0 + dx + 1;

        for (int cb = 0; cb < 8; ++cb) {
            const int c0 = cb * 32;
            __syncthreads();                // prior chunk fully consumed

            // Stage A: 16 pixels x 32 ci = 64 x 16B chunks (threads 0..63)
            if (tid < 64) {
                int m    = tid >> 2;
                int koff = (tid & 3) * 8;
                const _Float16* srcA = pf + (gy * PADW + gxb + m) * 256 + c0 + koff;
                copy16_g2l(srcA, &As[m * 32 + koff]);
                __builtin_prefetch(srcA + 32, 0, 0);   // next c-chunk hint
            }
            // Stage B: 256 n x 32 ci = 1024 x 16B chunks (all 256 threads x 4)
            #pragma unroll
            for (int i = 0; i < 4; ++i) {
                int q    = tid + 256 * i;
                int n    = q >> 2;
                int koff = (q & 3) * 8;
                copy16_g2l(Wt + (tap * 256 + n) * 256 + c0 + koff,
                           &Bs[n * 32 + koff]);
            }
            async_fence();
            __syncthreads();

            // A fragment: 16-bit A 16x32 ISA layout; per lane two contiguous
            // 8-element runs: K = half*8 + [0,8) and K = 16 + half*8 + [0,8)
            AFrag a;
            #pragma unroll
            for (int j = 0; j < 16; ++j) {
                int kl = (j < 8) ? (half * 8 + j) : (16 + half * 8 + (j - 8));
                a.e[j] = As[lm * 32 + kl];
            }
            // B fragments: 32x16, lane -> N = lm, K = same split as A (half*8 + j, 16 + half*8 + j-8)
            AFrag b0, b1;
            #pragma unroll
            for (int j = 0; j < 16; ++j) {
                int kk = (j < 8) ? (half * 8 + j) : (16 + half * 8 + (j - 8));
                b0.e[j] = Bs[(n0 + lm) * 32 + kk];
                b1.e[j] = Bs[(n0 + 16 + lm) * 32 + kk];
            }
            acc0.v = __builtin_amdgcn_wmma_f32_16x16x32_f16(
                false, a.v, false, b0.v, (short)0, acc0.v, false, false);
            acc1.v = __builtin_amdgcn_wmma_f32_16x16x32_f16(
                false, a.v, false, b1.v, (short)0, acc1.v, false, false);
        }
    }

    // C/D layout: lane%16 = N, VGPR r -> M = r + 8*(lane/16). Bias + ReLU.
    #pragma unroll
    for (int s = 0; s < 2; ++s) {
        const int n  = n0 + s * 16 + lm;
        const float bv = bias[n];
        CFrag* acc = s ? &acc1 : &acc0;
        #pragma unroll
        for (int r = 0; r < 8; ++r) {
            int M = r + 8 * half;
            int pixel = y * GRID_W + x0 + M;
            float v = acc->e[r] + bv;
            rf[pixel * 256 + n] = (_Float16)fmaxf(v, 0.f);
        }
    }
}

// ---------------------------------------------------------------------------
// Kernel 3: 1x1 heads + anchors + decode + sigmoid. One block (64 thr) / pixel.
// ---------------------------------------------------------------------------
__global__ void rpn_heads_kernel(const _Float16* __restrict__ rf,
                                 const float* __restrict__ cls_w, const float* __restrict__ cls_b,
                                 const float* __restrict__ bbox_w, const float* __restrict__ bbox_b,
                                 float* __restrict__ scores, float* __restrict__ boxes) {
    __shared__ _Float16 ch[256];
    __shared__ float val[45];
    const int pixel = blockIdx.x;
    const int t = threadIdx.x;

    for (int c = t; c < 256; c += 64) ch[c] = rf[pixel * 256 + c];
    __syncthreads();

    if (t < 45) {
        const float* wrow = (t < 9) ? (cls_w + t * 256) : (bbox_w + (t - 9) * 256);
        float s = 0.f;
        for (int c = 0; c < 256; ++c) s += wrow[c] * (float)ch[c];
        s += (t < 9) ? cls_b[t] : bbox_b[t - 9];
        val[t] = s;
    }
    __syncthreads();

    if (t < 9) {
        const int a = t;
        const float ar[3] = {0.5f, 1.0f, 2.0f};
        const float sc[3] = {128.f, 256.f, 512.f};
        int ai = a / 3, si = a % 3;
        float hr = sqrtf(ar[ai]);
        float hh = hr * sc[si];
        float ww = sc[si] / hr;
        float bx1 = rintf(-ww * 0.5f), by1 = rintf(-hh * 0.5f);
        float bx2 = rintf( ww * 0.5f), by2 = rintf( hh * 0.5f);
        int py = pixel / GRID_W, px = pixel % GRID_W;
        float sx = px * STRIDE_F, sy = py * STRIDE_F;
        float ax1 = sx + bx1, ay1 = sy + by1, ax2 = sx + bx2, ay2 = sy + by2;

        float aw = ax2 - ax1, ah = ay2 - ay1;
        float cx = ax1 + 0.5f * aw, cy = ay1 + 0.5f * ah;
        float dxv = val[9 + a * 4 + 0];
        float dyv = val[9 + a * 4 + 1];
        float dwv = fminf(val[9 + a * 4 + 2], DW_CLIPV);
        float dhv = fminf(val[9 + a * 4 + 3], DW_CLIPV);
        float pxc = dxv * aw + cx, pyc = dyv * ah + cy;
        float pw = expf(dwv) * aw, ph = expf(dhv) * ah;

        int i = pixel * 9 + a;
        boxes[i * 4 + 0] = pxc - 0.5f * pw;
        boxes[i * 4 + 1] = pyc - 0.5f * ph;
        boxes[i * 4 + 2] = pxc + 0.5f * pw;
        boxes[i * 4 + 3] = pyc + 0.5f * ph;
        scores[i] = 1.f / (1.f + expf(-val[a]));
    }
}

// ---------------------------------------------------------------------------
// Sorting helpers: u64 key = (~monotone(score)) << 32 | index  (asc sort ==
// descending score, ascending index tiebreak, matching stable top_k).
// ---------------------------------------------------------------------------
__device__ __forceinline__ u32 fkey(float s) {
    u32 u = __float_as_uint(s);
    return (u & 0x80000000u) ? ~u : (u | 0x80000000u);
}

__global__ void pack_keys_kernel(const float* __restrict__ scores, u64* __restrict__ keys,
                                 int n, int total) {
    int i = blockIdx.x * blockDim.x + threadIdx.x;
    if (i >= total) return;
    float s = (i < n) ? scores[i] : -INFINITY;
    keys[i] = ((u64)(~fkey(s)) << 32) | (u32)i;
}

__global__ void bitonic_step_kernel(u64* __restrict__ keys, int n, int k, int j) {
    int i = blockIdx.x * blockDim.x + threadIdx.x;
    if (i >= n) return;
    int ixj = i ^ j;
    if (ixj > i) {
        u64 a = keys[i], b = keys[ixj];
        bool asc = ((i & k) == 0);
        if ((a > b) == asc) { keys[i] = b; keys[ixj] = a; }
    }
}

// top-6000 gather + clip + min-size validity
__global__ void prep_topk_kernel(const u64* __restrict__ keys,
                                 const float* __restrict__ boxes,
                                 const float* __restrict__ scores,
                                 float* __restrict__ props, float* __restrict__ pscore,
                                 int* __restrict__ keep) {
    int t = blockIdx.x * blockDim.x + threadIdx.x;
    if (t >= PRENMS) return;
    int idx = (int)(keys[t] & 0xFFFFFFFFull);
    float x1 = fminf(fmaxf(boxes[idx * 4 + 0], 0.f), IM_SZ);
    float y1 = fminf(fmaxf(boxes[idx * 4 + 1], 0.f), IM_SZ);
    float x2 = fminf(fmaxf(boxes[idx * 4 + 2], 0.f), IM_SZ);
    float y2 = fminf(fmaxf(boxes[idx * 4 + 3], 0.f), IM_SZ);
    props[t * 4 + 0] = x1; props[t * 4 + 1] = y1;
    props[t * 4 + 2] = x2; props[t * 4 + 3] = y2;
    pscore[t] = scores[idx];
    keep[t] = ((x2 - x1) >= MIN_SZ) & ((y2 - y1) >= MIN_SZ);
}

// sequential-greedy NMS, one block, flags in LDS
__global__ void nms_kernel(const float* __restrict__ props, int* __restrict__ keep) {
    __shared__ unsigned char kp[PRENMS];
    const int t = threadIdx.x, nt = blockDim.x;
    for (int i = t; i < PRENMS; i += nt) kp[i] = (unsigned char)keep[i];
    __syncthreads();
    for (int i = 0; i < PRENMS; ++i) {
        if (kp[i]) {
            float bx1 = props[i * 4 + 0], by1 = props[i * 4 + 1];
            float bx2 = props[i * 4 + 2], by2 = props[i * 4 + 3];
            float areai = (bx2 - bx1) * (by2 - by1);
            for (int j = i + 1 + t; j < PRENMS; j += nt) {
                float x1 = fmaxf(bx1, props[j * 4 + 0]);
                float y1 = fmaxf(by1, props[j * 4 + 1]);
                float x2 = fminf(bx2, props[j * 4 + 2]);
                float y2 = fminf(by2, props[j * 4 + 3]);
                float inter = fmaxf(x2 - x1, 0.f) * fmaxf(y2 - y1, 0.f);
                float areaj = (props[j * 4 + 2] - props[j * 4 + 0]) *
                              (props[j * 4 + 3] - props[j * 4 + 1]);
                float iou = inter / (areai + areaj - inter);
                if (iou > NMS_THRV) kp[j] = 0;
            }
        }
        __syncthreads();
    }
    for (int i = t; i < PRENMS; i += nt) keep[i] = kp[i];
}

__global__ void pack_keys2_kernel(const float* __restrict__ pscore,
                                  const int* __restrict__ keep, u64* __restrict__ keys2) {
    int i = blockIdx.x * blockDim.x + threadIdx.x;
    if (i >= SORT2_N) return;
    float s = (i < PRENMS && keep[i]) ? pscore[i] : -INFINITY;
    keys2[i] = ((u64)(~fkey(s)) << 32) | (u32)i;
}

__global__ void finalize_kernel(const u64* __restrict__ keys2,
                                const float* __restrict__ props,
                                const float* __restrict__ pscore,
                                const int* __restrict__ keep,
                                float* __restrict__ out) {
    int t = blockIdx.x * blockDim.x + threadIdx.x;
    if (t >= POSTK) return;
    int idx = (int)(keys2[t] & 0xFFFFFFFFull);
    bool ok = (idx < PRENMS) && keep[idx];
    out[t * 5 + 0] = ok ? props[idx * 4 + 0] : 0.f;
    out[t * 5 + 1] = ok ? props[idx * 4 + 1] : 0.f;
    out[t * 5 + 2] = ok ? props[idx * 4 + 2] : 0.f;
    out[t * 5 + 3] = ok ? props[idx * 4 + 3] : 0.f;
    out[t * 5 + 4] = ok ? pscore[idx] : 0.f;
}

// ---------------------------------------------------------------------------
extern "C" void kernel_launch(void* const* d_in, const int* in_sizes, int n_in,
                              void* d_out, int out_size, void* d_ws, size_t ws_size,
                              hipStream_t stream) {
    const float* feat    = (const float*)d_in[1];
    const float* conv_w  = (const float*)d_in[2];
    const float* conv_b  = (const float*)d_in[3];
    const float* cls_w   = (const float*)d_in[4];
    const float* cls_b   = (const float*)d_in[5];
    const float* bbox_w  = (const float*)d_in[6];
    const float* bbox_b  = (const float*)d_in[7];

    char* ws = (char*)d_ws;
    size_t off = 0;
    _Float16* pf   = (_Float16*)(ws + off); off += (size_t)NPPIX * 256 * 2;     // 8.65 MB
    _Float16* Wt   = (_Float16*)(ws + off); off += (size_t)9 * 256 * 256 * 2;   // 1.18 MB
    _Float16* rf   = (_Float16*)(ws + off); off += (size_t)NPIX * 256 * 2;      // 8.39 MB
    float* scores  = (float*)(ws + off);    off += (size_t)N_ANCH * 4;          // 0.59 MB
    float* boxes   = (float*)(ws + off);    off += (size_t)N_ANCH * 16;         // 2.36 MB
    u64*  keys     = (u64*)(ws + off);      off += (size_t)SORT_N * 8;          // 2.10 MB
    float* props   = (float*)(ws + off);    off += (size_t)PRENMS * 16;
    float* pscore  = (float*)(ws + off);    off += (size_t)((PRENMS * 4 + 255) & ~255);
    int*  keep     = (int*)(ws + off);      off += (size_t)((PRENMS * 4 + 255) & ~255);
    u64*  keys2    = (u64*)(ws + off);      off += (size_t)SORT2_N * 8;

    feat_pad_kernel<<<(NPPIX * 256 + 255) / 256, 256, 0, stream>>>(feat, pf);
    rpn_wrepack_kernel<<<(9 * 256 * 256 + 255) / 256, 256, 0, stream>>>(conv_w, Wt);

    rpn_conv_wmma_kernel<<<1024, 256, 0, stream>>>(pf, Wt, conv_b, rf);

    rpn_heads_kernel<<<NPIX, 64, 0, stream>>>(rf, cls_w, cls_b, bbox_w, bbox_b,
                                              scores, boxes);

    pack_keys_kernel<<<SORT_N / 256, 256, 0, stream>>>(scores, keys, N_ANCH, SORT_N);
    for (int k = 2; k <= SORT_N; k <<= 1)
        for (int j = k >> 1; j > 0; j >>= 1)
            bitonic_step_kernel<<<SORT_N / 256, 256, 0, stream>>>(keys, SORT_N, k, j);

    prep_topk_kernel<<<(PRENMS + 255) / 256, 256, 0, stream>>>(keys, boxes, scores,
                                                               props, pscore, keep);
    nms_kernel<<<1, 1024, 0, stream>>>(props, keep);

    pack_keys2_kernel<<<SORT2_N / 256, 256, 0, stream>>>(pscore, keep, keys2);
    for (int k = 2; k <= SORT2_N; k <<= 1)
        for (int j = k >> 1; j > 0; j >>= 1)
            bitonic_step_kernel<<<SORT2_N / 256, 256, 0, stream>>>(keys2, SORT2_N, k, j);

    finalize_kernel<<<(POSTK + 255) / 256, 256, 0, stream>>>(keys2, props, pscore, keep,
                                                             (float*)d_out);
}
